// RotateLinear_2336462209049
// MI455X (gfx1250) — compile-verified, matrix operand order checked by reading, not run on hardware
//
#include <hip/hip_runtime.h>

// ---------------------------------------------------------------------------
// RotateLinear: out[8192,11008] = (FWHT(x)/sqrt(4096)) @ W^T + bias
//   1) fwht_rows_kernel : per-row FWHT in LDS, scale by 1/64, emit bf16 x_rot
//   2) w_to_bf16_kernel : convert W to bf16
//   3) gemm_wmma_kernel : 128x128x32 WMMA GEMM, 2-deep async global->LDS
//                         pipeline over 4 LDS buffers (ASYNCcnt-throttled)
// ---------------------------------------------------------------------------

typedef __attribute__((ext_vector_type(16))) __bf16 v16bf;
typedef __attribute__((ext_vector_type(8)))  float  v8f;

#define DIN   4096
#define DOUT  11008
#define MTOT  8192
#define BM    128
#define BN    128
#define BKS   32          // K step (matches WMMA K for bf16)
#define BKP   40          // padded LDS row stride in elements (32 + 8)
#define NBUF  4           // LDS pipeline depth (2 tiles in flight + compute)
#define BUFB  (BM * BKP * 2)   // bytes per buffer (10240)

union Frag { int4 i4[2]; v16bf v; };

// CDNA5 async global->LDS DMA (ASYNCcnt-tracked), GVS addressing.
__device__ __forceinline__ void async_ld_b128(unsigned lds, unsigned voff,
                                              unsigned long long base)
{
    asm volatile("global_load_async_to_lds_b128 %0, %1, %2"
                 :: "v"(lds), "v"(voff), "s"(base) : "memory");
}
__device__ __forceinline__ void wait_async0()
{
    asm volatile("s_wait_asynccnt 0" ::: "memory");
}
__device__ __forceinline__ void wait_async4()   // retire oldest tile only
{
    asm volatile("s_wait_asynccnt 4" ::: "memory");
}

// ---------------------------------------------------------------------------
// 1) Row-wise normalized fast Walsh-Hadamard transform, f32 -> bf16
// ---------------------------------------------------------------------------
__global__ __launch_bounds__(256)
void fwht_rows_kernel(const float* __restrict__ x, __bf16* __restrict__ xr)
{
    __shared__ float row[DIN];
    const int r   = blockIdx.x;            // 0..8191
    const int tid = threadIdx.x;

    const float* src = x + (size_t)r * DIN;
    for (int i = tid; i < DIN; i += 256) row[i] = src[i];
    __syncthreads();

    for (int s = 0; s < 12; ++s) {         // 4096 = 2^12
        const int h = 1 << s;
        for (int j = tid; j < DIN / 2; j += 256) {
            const int i0 = ((j >> s) << (s + 1)) | (j & (h - 1));
            const int i1 = i0 + h;
            const float a = row[i0];
            const float b = row[i1];
            row[i0] = a + b;
            row[i1] = a - b;
        }
        __syncthreads();
    }

    __bf16* dst = xr + (size_t)r * DIN;
    for (int i = tid; i < DIN; i += 256)
        dst[i] = (__bf16)(row[i] * 0.015625f);   // 1/sqrt(4096)
}

// ---------------------------------------------------------------------------
// 2) Weight f32 -> bf16
// ---------------------------------------------------------------------------
__global__ __launch_bounds__(256)
void w_to_bf16_kernel(const float4* __restrict__ w4,
                      uint2* __restrict__ wb4, long n4)
{
    long i            = (long)blockIdx.x * 256 + threadIdx.x;
    const long stride = (long)gridDim.x * 256;
    for (; i < n4; i += stride) {
        const float4 f = w4[i];
        union { __bf16 h[4]; uint2 u; } o;
        o.h[0] = (__bf16)f.x; o.h[1] = (__bf16)f.y;
        o.h[2] = (__bf16)f.z; o.h[3] = (__bf16)f.w;
        wb4[i] = o.u;
    }
}

// ---------------------------------------------------------------------------
// 3) WMMA GEMM: out[M,N] = A[M,K](bf16) * B[N,K](bf16)^T + bias
//    256 threads = 8 waves (wave32); block tile 128x128, K step 32.
//    Wave grid 2(M) x 4(N): each wave -> 64x32 = 4x2 WMMA 16x16 tiles.
//    4-buffer LDS ring, 2 tiles in flight via global_load_async_to_lds_b128.
// ---------------------------------------------------------------------------
__global__ __launch_bounds__(256)
void gemm_wmma_kernel(const __bf16* __restrict__ A,   // [MTOT][DIN]
                      const __bf16* __restrict__ Bm,  // [DOUT][DIN]
                      const float*  __restrict__ bias,
                      float* __restrict__ out)        // [MTOT][DOUT]
{
    __shared__ __bf16 lA[NBUF][BM * BKP];
    __shared__ __bf16 lB[NBUF][BN * BKP];

    const int tid   = threadIdx.x;
    const int lane  = tid & 31;
    const int wid   = tid >> 5;      // wave id 0..7 (wave32)
    const int waveM = wid >> 2;      // 0..1
    const int waveN = wid & 3;       // 0..3
    const int lmod  = lane & 15;
    const int lhi   = lane >> 4;     // 0/1: K half-group this lane owns

    const int rowBase = blockIdx.y * BM;
    const int colBase = blockIdx.x * BN;

    // Tile staging: 128 rows x 32 bf16 = 512 int4 per matrix; 256 threads ->
    // 2 async b128 per thread per matrix. Thread t covers rows r0 and r0+64.
    const unsigned r0 = (unsigned)tid >> 2;
    const unsigned kc = (unsigned)tid & 3;

    unsigned voffA = (((unsigned)rowBase + r0) * (unsigned)DIN + kc * 8u) * 2u;
    unsigned voffB = (((unsigned)colBase + r0) * (unsigned)DIN + kc * 8u) * 2u;
    const unsigned GROWS = 64u * DIN * 2u;                 // +64 rows (global)

    const unsigned ldsElem = (r0 * BKP + kc * 8u) * 2u;
    const unsigned ldsA0 = (unsigned)(uintptr_t)(&lA[0][0]) + ldsElem;
    const unsigned ldsB0 = (unsigned)(uintptr_t)(&lB[0][0]) + ldsElem;
    const unsigned LROWS = 64u * BKP * 2u;                 // +64 rows (LDS)

    const unsigned long long baseA = (unsigned long long)(uintptr_t)A;
    const unsigned long long baseB = (unsigned long long)(uintptr_t)Bm;

    v8f acc[4][2] = {};
    const int nk = DIN / BKS;        // 128 K tiles

    // -------- prologue: put tiles 0 and 1 in flight --------
    #pragma unroll
    for (int t = 0; t < 2; ++t) {
        const unsigned bo = (unsigned)t * BUFB;
        async_ld_b128(ldsA0 + bo,         voffA,         baseA);
        async_ld_b128(ldsA0 + bo + LROWS, voffA + GROWS, baseA);
        async_ld_b128(ldsB0 + bo,         voffB,         baseB);
        async_ld_b128(ldsB0 + bo + LROWS, voffB + GROWS, baseB);
        voffA += BKS * 2u; voffB += BKS * 2u;
    }

    for (int kt = 0; kt < nk; ++kt) {
        // Retire oldest tile (kt). In-order async completion: with tiles
        // kt,kt+1 outstanding (8 ops), cnt<=4 means tile kt has landed.
        if (kt + 1 < nk) wait_async4(); else wait_async0();

        // Issue tile kt+2 (buffer last read at kt-2; the kt-1 barrier,
        // already passed by every wave, guarantees those reads are done).
        if (kt + 2 < nk) {
            const unsigned bo = ((unsigned)(kt + 2) & (NBUF - 1)) * BUFB;
            async_ld_b128(ldsA0 + bo,         voffA,         baseA);
            async_ld_b128(ldsA0 + bo + LROWS, voffA + GROWS, baseA);
            async_ld_b128(ldsB0 + bo,         voffB,         baseB);
            async_ld_b128(ldsB0 + bo + LROWS, voffB + GROWS, baseB);
            voffA += BKS * 2u; voffB += BKS * 2u;
        }

        __syncthreads();      // all waves' tile-kt data visible in LDS

        // ---- compute: 4x2 WMMA tiles over this 32-wide K slab ----
        const int cur = kt & (NBUF - 1);
        Frag a[4], b[2];
        const int cb0 = lhi * 8;  // lanes 0-15: K 0..7,16..23 ; 16-31: 8..15,24..31
        #pragma unroll
        for (int mt = 0; mt < 4; ++mt) {
            const __bf16* base =
                &lA[cur][(waveM * 64 + mt * 16 + lmod) * BKP + cb0];
            a[mt].i4[0] = *(const int4*)(base);
            a[mt].i4[1] = *(const int4*)(base + 16);
        }
        #pragma unroll
        for (int nt = 0; nt < 2; ++nt) {
            const __bf16* base =
                &lB[cur][(waveN * 32 + nt * 16 + lmod) * BKP + cb0];
            b[nt].i4[0] = *(const int4*)(base);
            b[nt].i4[1] = *(const int4*)(base + 16);
        }
        #pragma unroll
        for (int mt = 0; mt < 4; ++mt)
            #pragma unroll
            for (int nt = 0; nt < 2; ++nt)
                acc[mt][nt] = __builtin_amdgcn_wmma_f32_16x16x32_bf16(
                    false, a[mt].v, false, b[nt].v,
                    (short)0, acc[mt][nt], false, false);
    }

    // -------- epilogue: bias add + store --------
    // C/D layout: VGPR i -> (M = i or i+8 by lane-half, N = lane%16)
    #pragma unroll
    for (int nt = 0; nt < 2; ++nt) {
        const int col = colBase + waveN * 32 + nt * 16 + lmod;
        const float bv = bias[col];
        #pragma unroll
        for (int mt = 0; mt < 4; ++mt) {
            #pragma unroll
            for (int i = 0; i < 8; ++i) {
                const int row = rowBase + waveM * 64 + mt * 16 + lhi * 8 + i;
                out[(size_t)row * DOUT + col] = acc[mt][nt][i] + bv;
            }
        }
    }
}

// ---------------------------------------------------------------------------
// Host launcher
// ---------------------------------------------------------------------------
extern "C" void kernel_launch(void* const* d_in, const int* in_sizes, int n_in,
                              void* d_out, int out_size, void* d_ws, size_t ws_size,
                              hipStream_t stream)
{
    const float* x    = (const float*)d_in[0];   // [4,2048,4096]
    const float* w    = (const float*)d_in[1];   // [11008,4096]
    const float* bias = (const float*)d_in[2];   // [11008]
    float* out        = (float*)d_out;           // [8192,11008]

    // workspace: x_rot bf16 (64 MiB) | W bf16 (86 MiB)
    __bf16* xr = (__bf16*)d_ws;
    __bf16* wb = (__bf16*)((char*)d_ws + (size_t)MTOT * DIN * sizeof(__bf16));

    fwht_rows_kernel<<<MTOT, 256, 0, stream>>>(x, xr);

    const long n4 = (long)DOUT * DIN / 4;
    w_to_bf16_kernel<<<4096, 256, 0, stream>>>((const float4*)w, (uint2*)wb, n4);

    dim3 grid(DOUT / BN, MTOT / BM);   // (86, 64)
    gemm_wmma_kernel<<<grid, 256, 0, stream>>>(xr, wb, bias, out);
}
